// VortexAttentionHead_87333864997589
// MI455X (gfx1250) — matrix operand order repaired and checked by reading, not compiled, not run on hardware
//
#include <hip/hip_runtime.h>
#include <hip/hip_bf16.h>

// ---------------------------------------------------------------------------
// Types for CDNA5 WMMA (gfx1250, wave32)
// ---------------------------------------------------------------------------
typedef __attribute__((ext_vector_type(16))) __bf16 bf16x16;
typedef __attribute__((ext_vector_type(8)))  float  f32x8;

#if __has_builtin(__builtin_amdgcn_sched_group_barrier)
#define SGB(mask, n) __builtin_amdgcn_sched_group_barrier((mask), (n), 0)
#else
#define SGB(mask, n)
#endif
// masks: 0x020 = VMEM read, 0x008 = MFMA/WMMA, 0x100 = DS read

union Frag {
    bf16x16 v;
    int4 q[2];
    unsigned short h[16];
};

__device__ __forceinline__ unsigned short f2bf(float x) {
    unsigned int u = __builtin_bit_cast(unsigned int, x);
    u += 0x7FFFu + ((u >> 16) & 1u);           // round-to-nearest-even
    return (unsigned short)(u >> 16);
}
__device__ __forceinline__ float bf2f(unsigned short h) {
    unsigned int u = ((unsigned int)h) << 16;
    return __builtin_bit_cast(float, u);
}

__device__ __forceinline__ f32x8 wmma_bf16(bf16x16 a, bf16x16 b, f32x8 c) {
    return __builtin_amdgcn_wmma_f32_16x16x32_bf16(false, a, false, b,
                                                   (short)0, c, false, false);
}

// A fragment (16x32 bf16, row-major source).
__device__ __forceinline__ bf16x16 load_a_bf16(const unsigned short* base, int row,
                                               int ldk, int d0, int lane) {
    int hh = lane >> 4;
    const unsigned short* p = base + (size_t)row * ldk + d0 + hh * 8;
    Frag f;
    f.q[0] = *(const int4*)(p);
    f.q[1] = *(const int4*)(p + 16);
    return f.v;
}

// A fragment built from f32 source (convert on the fly).
__device__ __forceinline__ bf16x16 load_a_f32(const float* base, int row,
                                              int ldk, int d0, int lane) {
    int hh = lane >> 4;
    const float* p = base + (size_t)row * ldk + d0 + hh * 8;
    float a0[8], a1[8];
    *(float4*)(a0)     = *(const float4*)(p);
    *(float4*)(a0 + 4) = *(const float4*)(p + 4);
    *(float4*)(a1)     = *(const float4*)(p + 16);
    *(float4*)(a1 + 4) = *(const float4*)(p + 20);
    Frag f;
#pragma unroll
    for (int i = 0; i < 8; i++) { f.h[i] = f2bf(a0[i]); f.h[8 + i] = f2bf(a1[i]); }
    return f.v;
}

// B fragment (32x16): B[k,n] = src[n0+n][d0+k] for row-major src [N,ldk].
__device__ __forceinline__ bf16x16 load_b_bf16(const unsigned short* base, int n0,
                                               int ldk, int d0, int lane) {
    int hh = lane >> 4, l = lane & 15;
    const unsigned short* p = base + (size_t)(n0 + l) * ldk + d0 + hh * 8;
    Frag f;
    f.q[0] = *(const int4*)(p);
    f.q[1] = *(const int4*)(p + 16);
    return f.v;
}

static __device__ const float INV_SCALE = 0.08838834764831845f; // 1/sqrt(128)

// ---------------------------------------------------------------------------
// Kernel: f32 -> bf16 conversion (weights)
// ---------------------------------------------------------------------------
__global__ void k_f2bf(const float* __restrict__ in, unsigned short* __restrict__ out, int n) {
    int i = blockIdx.x * 256 + threadIdx.x;
    if (i < n) out[i] = f2bf(in[i]);
}

// ---------------------------------------------------------------------------
// Kernel: fused QKV projection GEMMs. grid=(256,3); 128 threads (4 waves).
// Y[16384x128] = X[16384x1024] @ W^T.  Software-pipelined (double buffer).
// ---------------------------------------------------------------------------
__global__ __launch_bounds__(128) void k_proj3(const float* __restrict__ Xq,
                                               const float* __restrict__ Xk,
                                               const float* __restrict__ Xv,
                                               const unsigned short* __restrict__ Wqb,
                                               const unsigned short* __restrict__ Wkb,
                                               const unsigned short* __restrict__ Wvb,
                                               unsigned short* __restrict__ Yq,
                                               unsigned short* __restrict__ Yk,
                                               unsigned short* __restrict__ Yv) {
    const float* X = (blockIdx.y == 0) ? Xq : (blockIdx.y == 1) ? Xk : Xv;
    const unsigned short* Wb = (blockIdx.y == 0) ? Wqb : (blockIdx.y == 1) ? Wkb : Wvb;
    unsigned short* Yb = (blockIdx.y == 0) ? Yq : (blockIdx.y == 1) ? Yk : Yv;

    int lane = threadIdx.x & 31, w = threadIdx.x >> 5;
    int m0 = blockIdx.x * 64 + w * 16;
    int arow = m0 + (lane & 15);
    f32x8 zero = {0, 0, 0, 0, 0, 0, 0, 0};
    f32x8 acc[8];
#pragma unroll
    for (int nt = 0; nt < 8; nt++) acc[nt] = zero;

    bf16x16 b0[8], b1[8];
    bf16x16 a0, a1;
#pragma unroll
    for (int nt = 0; nt < 8; nt++) b0[nt] = load_b_bf16(Wb, nt * 16, 1024, 0, lane);
    a0 = load_a_f32(X, arow, 1024, 0, lane);

    for (int d0 = 0; d0 < 1024; d0 += 64) {
        // issue loads for d0+32, compute with buffer 0
#pragma unroll
        for (int nt = 0; nt < 8; nt++) b1[nt] = load_b_bf16(Wb, nt * 16, 1024, d0 + 32, lane);
        a1 = load_a_f32(X, arow, 1024, d0 + 32, lane);
#pragma unroll
        for (int nt = 0; nt < 8; nt++) acc[nt] = wmma_bf16(a0, b0[nt], acc[nt]);
        SGB(0x020, 20); SGB(0x008, 8);
        // issue loads for d0+64, compute with buffer 1
        if (d0 + 64 < 1024) {
#pragma unroll
            for (int nt = 0; nt < 8; nt++)
                b0[nt] = load_b_bf16(Wb, nt * 16, 1024, d0 + 64, lane);
            a0 = load_a_f32(X, arow, 1024, d0 + 64, lane);
        }
#pragma unroll
        for (int nt = 0; nt < 8; nt++) acc[nt] = wmma_bf16(a1, b1[nt], acc[nt]);
        SGB(0x020, 20); SGB(0x008, 8);
    }
    int hh = lane >> 4, l = lane & 15;
#pragma unroll
    for (int nt = 0; nt < 8; nt++)
#pragma unroll
        for (int r = 0; r < 8; r++)
            Yb[(size_t)(m0 + 8 * hh + r) * 128 + nt * 16 + l] = f2bf(acc[nt][r]);
}

// ---------------------------------------------------------------------------
// Kernel: pol/tor GEMMs + vortex value blend -> cv (f32)
// ---------------------------------------------------------------------------
__global__ __launch_bounds__(128) void k_polvor(const unsigned short* __restrict__ Vb,
                                                const unsigned short* __restrict__ Wp,
                                                const unsigned short* __restrict__ Wt,
                                                float* __restrict__ cv) {
    int lane = threadIdx.x & 31, w = threadIdx.x >> 5;
    int m0 = blockIdx.x * 64 + w * 16;
    int arow = m0 + (lane & 15);
    f32x8 zero = {0, 0, 0, 0, 0, 0, 0, 0};
    f32x8 ap[8], at[8];
#pragma unroll
    for (int nt = 0; nt < 8; nt++) { ap[nt] = zero; at[nt] = zero; }

#pragma unroll
    for (int d0 = 0; d0 < 128; d0 += 32) {
        bf16x16 a = load_a_bf16(Vb, arow, 128, d0, lane);
        bf16x16 bb[8];
#pragma unroll
        for (int nt = 0; nt < 8; nt++) bb[nt] = load_b_bf16(Wp, nt * 16, 128, d0, lane);
#pragma unroll
        for (int nt = 0; nt < 8; nt++) ap[nt] = wmma_bf16(a, bb[nt], ap[nt]);
        SGB(0x020, 18); SGB(0x008, 8);
#pragma unroll
        for (int nt = 0; nt < 8; nt++) bb[nt] = load_b_bf16(Wt, nt * 16, 128, d0, lane);
#pragma unroll
        for (int nt = 0; nt < 8; nt++) at[nt] = wmma_bf16(a, bb[nt], at[nt]);
        SGB(0x020, 16); SGB(0x008, 8);
    }
    int hh = lane >> 4, l = lane & 15;
#pragma unroll
    for (int nt = 0; nt < 8; nt++)
#pragma unroll
        for (int r = 0; r < 8; r++) {
            int row = m0 + 8 * hh + r, col = nt * 16 + l;
            float vv = bf2f(Vb[(size_t)row * 128 + col]);
            cv[(size_t)row * 128 + col] = 0.2f * vv + 0.48f * ap[nt][r] + 0.32f * at[nt][r];
        }
}

// ---------------------------------------------------------------------------
// Kernel: cv2 (shift-folded values), stored TRANSPOSED: cv2T[b][h][j] (bf16)
// ---------------------------------------------------------------------------
__global__ void k_cv2(const float* __restrict__ cv, unsigned short* __restrict__ cv2T) {
    size_t i = (size_t)blockIdx.x * 256 + threadIdx.x; // i = ((b*128+h)*4096 + j)
    if (i >= (size_t)4 * 128 * 4096) return;
    int j = (int)(i & 4095);
    size_t bh = i >> 12;
    int h = (int)(bh & 127);
    int b = (int)(bh >> 7);
    int j1 = (j + 1) & 4095, j2 = (j + 1024) & 4095;
    const float* base = cv + (size_t)b * 4096 * 128 + h;
    float r = 0.3f * base[(size_t)j * 128] + 0.49f * base[(size_t)j1 * 128] +
              0.21f * base[(size_t)j2 * 128];
    cv2T[i] = f2bf(r);
}

// ---------------------------------------------------------------------------
// Kernel: pass1 — flash-style per-row softmax stats (m, l). 256 blocks, 4 waves.
// Rotating preload: next tile-pair's loads issue right after current WMMAs,
// covered by the per-k0 reduction epilogue.
// ---------------------------------------------------------------------------
__global__ __launch_bounds__(128) void k_pass1(const unsigned short* __restrict__ Qb,
                                               const unsigned short* __restrict__ Kb,
                                               float* __restrict__ Mrow,
                                               float* __restrict__ Lrow) {
    int lane = threadIdx.x & 31, w = threadIdx.x >> 5;
    int bq = blockIdx.x, b = bq >> 6, qt = bq & 63;
    int qrow0 = b * 4096 + qt * 64 + w * 16;
    int arow = qrow0 + (lane & 15);
    int kbase = b * 4096;

    bf16x16 aq[4];
#pragma unroll
    for (int ks = 0; ks < 4; ks++) aq[ks] = load_a_bf16(Qb, arow, 128, ks * 32, lane);

    float m[8], l[8];
#pragma unroll
    for (int r = 0; r < 8; r++) { m[r] = -1e30f; l[r] = 0.0f; }

    f32x8 zero = {0, 0, 0, 0, 0, 0, 0, 0};
    bf16x16 bA[4], bB[4];
#pragma unroll
    for (int ks = 0; ks < 4; ks++) {
        bA[ks] = load_b_bf16(Kb, kbase + 0, 128, ks * 32, lane);
        bB[ks] = load_b_bf16(Kb, kbase + 16, 128, ks * 32, lane);
    }

    for (int k0 = 0; k0 < 4096; k0 += 128) {
        f32x8 s[8];
#pragma unroll
        for (int tp = 0; tp < 4; tp++) {
            f32x8 s0 = zero, s1 = zero;
#pragma unroll
            for (int ks = 0; ks < 4; ks++) {
                s0 = wmma_bf16(aq[ks], bA[ks], s0);
                s1 = wmma_bf16(aq[ks], bB[ks], s1);
            }
            SGB(0x008, 8);
            // rotate: load next tile pair (wraps over whole key range)
            int kn = (k0 + tp * 32 + 32) & 4095;
#pragma unroll
            for (int ks = 0; ks < 4; ks++) {
                bA[ks] = load_b_bf16(Kb, kbase + kn, 128, ks * 32, lane);
                bB[ks] = load_b_bf16(Kb, kbase + kn + 16, 128, ks * 32, lane);
            }
            SGB(0x020, 16);
            s[tp * 2] = s0;
            s[tp * 2 + 1] = s1;
        }
#pragma unroll
        for (int r = 0; r < 8; r++) {
            float mx = -1e30f;
#pragma unroll
            for (int tt = 0; tt < 8; tt++) mx = fmaxf(mx, s[tt][r] * INV_SCALE);
            mx = fmaxf(mx, __shfl_xor(mx, 1, 16));
            mx = fmaxf(mx, __shfl_xor(mx, 2, 16));
            mx = fmaxf(mx, __shfl_xor(mx, 4, 16));
            mx = fmaxf(mx, __shfl_xor(mx, 8, 16));
            float mn = fmaxf(m[r], mx);
            float sum = 0.0f;
#pragma unroll
            for (int tt = 0; tt < 8; tt++) sum += __expf(s[tt][r] * INV_SCALE - mn);
            sum += __shfl_xor(sum, 1, 16);
            sum += __shfl_xor(sum, 2, 16);
            sum += __shfl_xor(sum, 4, 16);
            sum += __shfl_xor(sum, 8, 16);
            l[r] = l[r] * __expf(m[r] - mn) + sum;
            m[r] = mn;
        }
    }
    if ((lane & 15) == 0) {
        int rbase = qrow0 + 8 * (lane >> 4);
#pragma unroll
        for (int r = 0; r < 8; r++) { Mrow[rbase + r] = m[r]; Lrow[rbase + r] = l[r]; }
    }
}

// ---------------------------------------------------------------------------
// Kernel: pass2 — write vortex_attn once, accumulate out = attn @ cv2T
// 256 blocks, 4 waves, 16 KB LDS P tile. Rotating preload: next subtile's
// loads issue after current WMMAs, covered by the exp/shfl/store epilogue.
// ---------------------------------------------------------------------------
__global__ __launch_bounds__(128) void k_pass2(const unsigned short* __restrict__ Qb,
                                               const unsigned short* __restrict__ Kb,
                                               const unsigned short* __restrict__ cv2T,
                                               const float* __restrict__ Mrow,
                                               const float* __restrict__ Lrow,
                                               float* __restrict__ vout,
                                               float* __restrict__ oattn) {
    __shared__ unsigned short P[64 * 128]; // 16 KB normalized attn tile (bf16)
    int lane = threadIdx.x & 31, w = threadIdx.x >> 5;
    int hh = lane >> 4, l = lane & 15;
    int bq = blockIdx.x, b = bq >> 6, qt = bq & 63;
    int qrow0 = b * 4096 + qt * 64 + w * 16; // global row in [B*S]
    int arow = qrow0 + l;
    int kbase = b * 4096;

    bf16x16 aq[4];
#pragma unroll
    for (int ks = 0; ks < 4; ks++) aq[ks] = load_a_bf16(Qb, arow, 128, ks * 32, lane);

    float rm[8], rinv[8];
#pragma unroll
    for (int r = 0; r < 8; r++) {
        int rr = qrow0 + 8 * hh + r;
        rm[r] = Mrow[rr];
        rinv[r] = 1.0f / Lrow[rr];
    }

    const unsigned short* cvb = cv2T + (size_t)b * 128 * 4096;
    f32x8 zero = {0, 0, 0, 0, 0, 0, 0, 0};
    f32x8 oacc[8];
#pragma unroll
    for (int nt = 0; nt < 8; nt++) oacc[nt] = zero;

    // prevp: normalized attn at column (k-1) wrap — init from keys 4080..4095
    float prevp[8];
    {
        bf16x16 binit[4];
#pragma unroll
        for (int ks = 0; ks < 4; ks++)
            binit[ks] = load_b_bf16(Kb, kbase + 4080, 128, ks * 32, lane);
        f32x8 s = zero;
#pragma unroll
        for (int ks = 0; ks < 4; ks++) s = wmma_bf16(aq[ks], binit[ks], s);
#pragma unroll
        for (int r = 0; r < 8; r++) {
            float p = __expf(s[r] * INV_SCALE - rm[r]) * rinv[r];
            prevp[r] = __shfl(p, 15, 16);
        }
    }

    // rotating fragment buffers, preloaded for subtile kc=0
    bf16x16 bmn[4], bfr[4];
#pragma unroll
    for (int ks = 0; ks < 4; ks++) {
        bmn[ks] = load_b_bf16(Kb, kbase + 0, 128, ks * 32, lane);
        bfr[ks] = load_b_bf16(Kb, kbase + 3072, 128, ks * 32, lane);
    }

    for (int k0 = 0; k0 < 4096; k0 += 128) {
        __builtin_prefetch(Kb + (size_t)(kbase + ((k0 + 128) & 4095)) * 128, 0, 1);
#pragma unroll 1
        for (int tt = 0; tt < 8; tt++) {
            int kc = k0 + tt * 16;
            f32x8 s = zero, sf = zero;
#pragma unroll
            for (int ks = 0; ks < 4; ks++) {
                s  = wmma_bf16(aq[ks], bmn[ks], s);
                sf = wmma_bf16(aq[ks], bfr[ks], sf);
            }
            SGB(0x008, 8);
            // rotate: issue loads for the next subtile (wraps across k0 too)
            int kcn = (kc + 16) & 4095;
            int kfn = (kcn + 3072) & 4095;
#pragma unroll
            for (int ks = 0; ks < 4; ks++) {
                bmn[ks] = load_b_bf16(Kb, kbase + kcn, 128, ks * 32, lane);
                bfr[ks] = load_b_bf16(Kb, kbase + kfn, 128, ks * 32, lane);
            }
            SGB(0x020, 16);
            // epilogue: normalize, vortex blend, store, stage P (covers loads)
#pragma unroll
            for (int r = 0; r < 8; r++) {
                float p  = __expf(s[r]  * INV_SCALE - rm[r]) * rinv[r];
                float pf = __expf(sf[r] * INV_SCALE - rm[r]) * rinv[r];
                float pm1 = __shfl_up(p, 1, 16);
                if (l == 0) pm1 = prevp[r];
                float vx = 0.3f * p + 0.49f * pm1 + 0.21f * pf;
                int rowl = qt * 64 + w * 16 + 8 * hh + r;
                vout[((size_t)(b * 4096 + rowl)) * 4096 + kc + l] = vx;
                prevp[r] = __shfl(p, 15, 16);
                P[(w * 16 + 8 * hh + r) * 128 + tt * 16 + l] = f2bf(p);
            }
        }
        __syncthreads();
        // out += P(16x128 rows of this wave) @ cv2T
#pragma unroll
        for (int ks2 = 0; ks2 < 4; ks2++) {
            bf16x16 bb[8];
#pragma unroll
            for (int nt = 0; nt < 8; nt++)
                bb[nt] = load_b_bf16(cvb, nt * 16, 4096, k0 + ks2 * 32, lane);
            bf16x16 ap_ = load_a_bf16(&P[0], w * 16 + l, 128, ks2 * 32, lane);
#pragma unroll
            for (int nt = 0; nt < 8; nt++)
                oacc[nt] = wmma_bf16(ap_, bb[nt], oacc[nt]);
            SGB(0x020, 16); SGB(0x100, 2); SGB(0x008, 8);
        }
        __syncthreads();
    }
#pragma unroll
    for (int nt = 0; nt < 8; nt++)
#pragma unroll
        for (int r = 0; r < 8; r++)
            oattn[(size_t)(qrow0 + 8 * hh + r) * 128 + nt * 16 + l] = oacc[nt][r];
}

// ---------------------------------------------------------------------------
// Kernel: sigmoid-gated memory fusion (K=256 WMMA GEMM + elementwise)
// ---------------------------------------------------------------------------
__global__ __launch_bounds__(128) void k_gate(const float* __restrict__ oattn,
                                              const float* __restrict__ prev,
                                              const unsigned short* __restrict__ Wmb,
                                              const float* __restrict__ bm,
                                              float* __restrict__ outp) {
    int lane = threadIdx.x & 31, w = threadIdx.x >> 5;
    int hh = lane >> 4, l = lane & 15;
    int m0 = blockIdx.x * 64 + w * 16;
    int arow = m0 + l;
    f32x8 zero = {0, 0, 0, 0, 0, 0, 0, 0};
    f32x8 acc[8];
#pragma unroll
    for (int nt = 0; nt < 8; nt++) acc[nt] = zero;

#pragma unroll
    for (int ks = 0; ks < 4; ks++) {
        bf16x16 bb[8];
#pragma unroll
        for (int nt = 0; nt < 8; nt++)
            bb[nt] = load_b_bf16(Wmb, nt * 16, 256, ks * 32, lane);
        bf16x16 a = load_a_f32(oattn, arow, 128, ks * 32, lane);
#pragma unroll
        for (int nt = 0; nt < 8; nt++) acc[nt] = wmma_bf16(a, bb[nt], acc[nt]);
        SGB(0x020, 20); SGB(0x008, 8);
    }
#pragma unroll
    for (int ks = 0; ks < 4; ks++) {
        bf16x16 bb[8];
#pragma unroll
        for (int nt = 0; nt < 8; nt++)
            bb[nt] = load_b_bf16(Wmb, nt * 16, 256, 128 + ks * 32, lane);
        bf16x16 a = load_a_f32(prev, arow, 128, ks * 32, lane);
#pragma unroll
        for (int nt = 0; nt < 8; nt++) acc[nt] = wmma_bf16(a, bb[nt], acc[nt]);
        SGB(0x020, 20); SGB(0x008, 8);
    }
#pragma unroll
    for (int nt = 0; nt < 8; nt++) {
        float bias = bm[nt * 16 + l];
#pragma unroll
        for (int r = 0; r < 8; r++) {
            int row = m0 + 8 * hh + r, col = nt * 16 + l;
            float g = 1.0f / (1.0f + __expf(-(acc[nt][r] + bias)));
            float pv = prev[(size_t)row * 128 + col];
            float ov = oattn[(size_t)row * 128 + col];
            outp[(size_t)row * 128 + col] = 0.9f * g * pv + 0.1f * ov;
        }
    }
}

// ---------------------------------------------------------------------------
// Host launcher
// ---------------------------------------------------------------------------
extern "C" void kernel_launch(void* const* d_in, const int* in_sizes, int n_in,
                              void* d_out, int out_size, void* d_ws, size_t ws_size,
                              hipStream_t stream) {
    (void)in_sizes; (void)n_in; (void)out_size; (void)ws_size;
    const float* query = (const float*)d_in[0];
    const float* key   = (const float*)d_in[1];
    const float* value = (const float*)d_in[2];
    const float* prevm = (const float*)d_in[3];
    const float* Wq = (const float*)d_in[4];
    const float* Wk = (const float*)d_in[5];
    const float* Wv = (const float*)d_in[6];
    const float* Wp = (const float*)d_in[7];
    const float* Wt = (const float*)d_in[8];
    const float* Wm = (const float*)d_in[9];
    const float* bm = (const float*)d_in[10];

    const size_t BS = 4 * 4096;          // 16384 rows
    char* ws = (char*)d_ws;
    size_t off = 0;
    auto alloc = [&](size_t bytes) -> void* {
        void* p = ws + off;
        off = (off + bytes + 255) & ~(size_t)255;
        return p;
    };
    unsigned short* Qb   = (unsigned short*)alloc(BS * 128 * 2);
    unsigned short* Kb   = (unsigned short*)alloc(BS * 128 * 2);
    unsigned short* Vb   = (unsigned short*)alloc(BS * 128 * 2);
    unsigned short* Wqb  = (unsigned short*)alloc(128 * 1024 * 2);
    unsigned short* Wkb  = (unsigned short*)alloc(128 * 1024 * 2);
    unsigned short* Wvb  = (unsigned short*)alloc(128 * 1024 * 2);
    unsigned short* Wpb  = (unsigned short*)alloc(128 * 128 * 2);
    unsigned short* Wtb  = (unsigned short*)alloc(128 * 128 * 2);
    unsigned short* Wmb  = (unsigned short*)alloc(128 * 256 * 2);
    float*          cv   = (float*)alloc(BS * 128 * 4);
    unsigned short* cv2T = (unsigned short*)alloc(BS * 128 * 2);
    float*          Mrow = (float*)alloc(BS * 4);
    float*          Lrow = (float*)alloc(BS * 4);
    float*          oatt = (float*)alloc(BS * 128 * 4);

    float* out_final = (float*)d_out;                       // [4,4096,128]
    float* vout      = (float*)d_out + (size_t)BS * 128;    // [4,4096,4096]

    // 1) weights -> bf16
    k_f2bf<<<(128 * 1024 + 255) / 256, 256, 0, stream>>>(Wq, Wqb, 128 * 1024);
    k_f2bf<<<(128 * 1024 + 255) / 256, 256, 0, stream>>>(Wk, Wkb, 128 * 1024);
    k_f2bf<<<(128 * 1024 + 255) / 256, 256, 0, stream>>>(Wv, Wvb, 128 * 1024);
    k_f2bf<<<(128 * 128 + 255) / 256, 256, 0, stream>>>(Wp, Wpb, 128 * 128);
    k_f2bf<<<(128 * 128 + 255) / 256, 256, 0, stream>>>(Wt, Wtb, 128 * 128);
    k_f2bf<<<(128 * 256 + 255) / 256, 256, 0, stream>>>(Wm, Wmb, 128 * 256);

    // 2) fused QKV projections: grid (256, 3)
    k_proj3<<<dim3(256, 3), 128, 0, stream>>>(query, key, value, Wqb, Wkb, Wvb,
                                              Qb, Kb, Vb);

    // 3) vortex value blend
    k_polvor<<<256, 128, 0, stream>>>(Vb, Wpb, Wtb, cv);

    // 4) shift-folded cv2 (transposed bf16)
    k_cv2<<<(int)((BS * 128 + 255) / 256), 256, 0, stream>>>(cv, cv2T);

    // 5) softmax stats
    k_pass1<<<256, 128, 0, stream>>>(Qb, Kb, Mrow, Lrow);

    // 6) vortex_attn write + out = attn @ cv2
    k_pass2<<<256, 128, 0, stream>>>(Qb, Kb, cv2T, Mrow, Lrow, vout, oatt);

    // 7) gated memory fusion
    k_gate<<<256, 128, 0, stream>>>(oatt, prevm, Wmb, bm, out_final);
}
